// MultiScaleDecoder_84430467105494
// MI455X (gfx1250) — compile-verified, matrix operand order checked by reading, not run on hardware
//
#include <hip/hip_runtime.h>
#include <hip/hip_bf16.h>

// ---------------- geometry ----------------
#define Bq    64
#define ZDq   128
#define EDq   256
#define HSq   1024
#define HFq   1024
#define Tq    512
#define NG    3072            // 3*H gate width
#define NTG   192             // NG/16 N-tiles
#define MTQ   4               // 64/16 M-tiles
#define KT_HH 32              // K=1024 -> 32 k-tiles of 32
#define KT_Z  4               // K=128
#define KT_E  8               // K=256
#define NSLOW 64              // T/K slow updates

#define NWG   96
#define TPB   256             // 8 waves of 32
#define ABLOB_BYTES (Bq * HFq * 2)   // 131072: one A-operand blob (h state)

typedef __attribute__((ext_vector_type(16))) __bf16 v16bf;
typedef __attribute__((ext_vector_type(8)))  float  v8f;
typedef unsigned int v4u  __attribute__((ext_vector_type(4)));
typedef int          v8i_ __attribute__((ext_vector_type(8)));
typedef int          v4i_ __attribute__((ext_vector_type(4)));

#ifndef __has_builtin
#define __has_builtin(x) 0
#endif
#if __has_builtin(__builtin_amdgcn_tensor_load_to_lds) && \
    __has_builtin(__builtin_amdgcn_s_wait_tensorcnt)
#define USE_TDM 1
#else
#define USE_TDM 0
#endif

__device__ __forceinline__ v8f wmma_bf16(v16bf a, v16bf b, v8f c) {
  // D = A(16x32 bf16) * B(32x16 bf16) + C(f32)
  return __builtin_amdgcn_wmma_f32_16x16x32_bf16(false, a, false, b, (short)0, c,
                                                 false, false);
}
__device__ __forceinline__ float sigmf(float x) { return 1.0f / (1.0f + __expf(-x)); }

// Broadcast one 128KB A-operand blob (64x1024 bf16, WMMA A layout) from global
// into this workgroup's LDS via the Tensor Data Mover (D# per CDNA5 ISA ch.8):
// 2-D tensor, data_size=8B, tensor_dim0=tile_dim0=16384 elems, tile_dim1=1.
__device__ __forceinline__ void stage_to_lds(const __bf16* gsrc, __bf16* lds,
                                             int wave) {
#if USE_TDM
  if (wave == 0) {
    unsigned long long ga = (unsigned long long)(const void*)gsrc;
    unsigned lds_off = (unsigned)(unsigned long long)(void*)lds;  // addr[31:0] = LDS byte offset
    v4u g0 = { 1u,                                   // count=1 (valid user D#)
               lds_off,                               // lds_addr
               (unsigned)(ga & 0xFFFFFFFFu),          // global_addr[31:0]
               (unsigned)((ga >> 32) & 0x1FFFFFFu) | (2u << 30) }; // addr[56:32] | type=2
    v8i_ g1 = { (int)0x00030000,   // wg_mask=0, data_size=3 (8B)
                (int)0x40000000,   // tensor_dim0[15:0]=16384 @ bits 63:48
                (int)0x00010000,   // tensor_dim0 hi=0, tensor_dim1=1 @ bits 95:80
                (int)0x40000000,   // tensor_dim1 hi=0, tile_dim0=16384 @ bits 127:112
                1,                 // tile_dim1=1, tile_dim2=0
                16384,             // tensor_dim0_stride
                0, 0 };
    v4i_ zz = { 0, 0, 0, 0 };
#if __clang_major__ >= 23
    v8i_ z8 = { 0, 0, 0, 0, 0, 0, 0, 0 };
    __builtin_amdgcn_tensor_load_to_lds(g0, g1, zz, zz, z8, 0);
#else
    __builtin_amdgcn_tensor_load_to_lds(g0, g1, zz, zz, 0);
#endif
    __builtin_amdgcn_s_wait_tensorcnt((short)0);
  }
  __syncthreads();
#else
  const ulonglong2* s = (const ulonglong2*)gsrc;
  ulonglong2* d = (ulonglong2*)lds;
  for (int i = threadIdx.x; i < ABLOB_BYTES / 16; i += TPB) d[i] = s[i];
  __syncthreads();
#endif
}

// ---------------- pre-pass: pack weight slice [N][ldw] cols [col0,col0+Kd) into
// WMMA B-operand bf16 tiles: tile(nt,kt) = 32 lanes x 16 bf16.
__global__ void msdec_pack_b(const float* __restrict__ w, int ldw, int col0,
                             int Kd, int Nd, __bf16* __restrict__ dst) {
  const int KT = Kd >> 5, NT = Nd >> 4;
  const int total = NT * KT * 32;
  for (int i = blockIdx.x * blockDim.x + threadIdx.x; i < total;
       i += gridDim.x * blockDim.x) {
    const int lane = i & 31;
    const int tile = i >> 5;
    const int kt   = tile % KT;
    const int ntl  = tile / KT;
    const int n    = ntl * 16 + (lane & 15);
    const int khalf = lane >> 4;
    __bf16* o = dst + (size_t)tile * 512 + (size_t)lane * 16;
    const float* wr = w + (size_t)n * ldw + col0 + kt * 32;
#pragma unroll
    for (int j = 0; j < 16; ++j) {
      int k = (j < 8) ? (khalf * 8 + j) : (16 + khalf * 8 + (j - 8));
      o[j] = (__bf16)wr[k];
    }
  }
}

// pack activation [M][ldx] into A-operand tiles (same lane->K mapping)
__global__ void msdec_pack_a(const float* __restrict__ x, int ldx, int Kd, int Md,
                             __bf16* __restrict__ dst) {
  const int KT = Kd >> 5, MT = Md >> 4;
  const int total = MT * KT * 32;
  for (int i = blockIdx.x * blockDim.x + threadIdx.x; i < total;
       i += gridDim.x * blockDim.x) {
    const int lane = i & 31;
    const int tile = i >> 5;
    const int kt   = tile % KT;
    const int mtl  = tile / KT;
    const int m    = mtl * 16 + (lane & 15);
    const int khalf = lane >> 4;
    __bf16* o = dst + (size_t)tile * 512 + (size_t)lane * 16;
    const float* xr = x + (size_t)m * ldx + kt * 32;
#pragma unroll
    for (int j = 0; j < 16; ++j) {
      int k = (j < 8) ? (khalf * 8 + j) : (16 + khalf * 8 + (j - 8));
      o[j] = (__bf16)xr[k];
    }
  }
}

// pack teacher-forced z_prev: ztp[t] = z[:, :, t-1] (zeros at t=0), A-layout per t.
__global__ void msdec_pack_ztp(const float* __restrict__ z, __bf16* __restrict__ dst) {
  const int total = Tq * 8192;
  for (int i = blockIdx.x * blockDim.x + threadIdx.x; i < total;
       i += gridDim.x * blockDim.x) {
    const int t    = i >> 13;
    const int e    = i & 8191;
    const int tile = e >> 9;     // mt*4 + kt
    const int r    = e & 511;
    const int kt   = tile & 3, mt = tile >> 2;
    const int lane = r >> 4, j = r & 15;
    const int b    = mt * 16 + (lane & 15);
    const int khalf = lane >> 4;
    const int d = kt * 32 + ((j < 8) ? (khalf * 8 + j) : (16 + khalf * 8 + (j - 8)));
    float v = (t == 0) ? 0.0f : z[((size_t)b * ZDq + d) * Tq + (t - 1)];
    dst[i] = (__bf16)v;
  }
}

__global__ void msdec_init(unsigned* bar, float* hF, float* hsF,
                           unsigned short* hA, unsigned short* hsA0) {
  const int n = Bq * HFq;  // 65536
  for (int i = blockIdx.x * blockDim.x + threadIdx.x; i < n;
       i += gridDim.x * blockDim.x) {
    hF[i] = 0.0f; hsF[i] = 0.0f; hA[i] = 0; hsA0[i] = 0;
  }
  if (blockIdx.x == 0 && threadIdx.x == 0) *bar = 0u;
}

// ---------------- persistent decoder kernel ----------------
struct MSDecParams {
  const float *sb_ih, *sb_hh, *fb_ih, *fb_hh, *mu_b, *lv_b;
  const __bf16 *WzB, *WhsB, *WeB, *WhhB, *SwihB, *SwhhB, *MuB, *LvB;
  const __bf16 *eA, *ztpA;
  __bf16 *hA, *hsA0, *hsSlots;           // hsSlots: 64 slots of 65536 bf16
  float *giE, *giS, *giBase, *gA, *gB;   // [64][3072] f32 each
  float *hF, *hsF;                       // [64][1024] f32
  float *out;                            // mu then logvar, [B][ZD][T]
  unsigned *bar;
};

__device__ __forceinline__ void gbar(unsigned* bar, unsigned g) {
  __threadfence();                       // release this block's stores
  __syncthreads();
  if (threadIdx.x == 0) {
    __hip_atomic_fetch_add(bar, 1u, __ATOMIC_RELEASE, __HIP_MEMORY_SCOPE_AGENT);
    while (__hip_atomic_load(bar, __ATOMIC_ACQUIRE, __HIP_MEMORY_SCOPE_AGENT) <
           g * (unsigned)NWG)
      __builtin_amdgcn_s_sleep(1);
  }
  __syncthreads();
  __threadfence();                       // acquire (invalidate stale near caches)
}

__global__ __launch_bounds__(TPB, 1) void msdec_main(MSDecParams p) {
  extern __shared__ __bf16 ldsA[];                   // 131072 B: staged A blob
  const int lane = threadIdx.x & 31;
  const int wave = threadIdx.x >> 5;
  const int gw   = blockIdx.x * (TPB / 32) + wave;   // 0..767  (one C-tile / wave)
  const int tid  = blockIdx.x * TPB + threadIdx.x;   // 0..24575
  const int mt   = gw & 3;                           // 4 M-tiles (B=64)
  const int nt   = gw >> 2;                          // 192 N-tiles (3072)
  const int nn   = nt * 16 + (lane & 15);
  const int mr0  = mt * 16 + (lane >> 4) * 8;        // C layout rows
  unsigned gen = 0;

  const v16bf* lA = (const v16bf*)ldsA + (size_t)mt * KT_HH * 32 + lane;

  // ---- phase 0: constant input-side gates from e (K=256) for fast & slow GRU
  {
    const v16bf* A  = (const v16bf*)p.eA    + (size_t)mt * KT_E * 32 + lane;
    const v16bf* B1 = (const v16bf*)p.WeB   + (size_t)nt * KT_E * 32 + lane;
    const v16bf* B2 = (const v16bf*)p.SwihB + (size_t)nt * KT_E * 32 + lane;
    v8f a1 = {}, a2 = {};
#pragma unroll
    for (int kt = 0; kt < KT_E; ++kt) {
      v16bf av = A[kt * 32];
      a1 = wmma_bf16(av, B1[kt * 32], a1);
      a2 = wmma_bf16(av, B2[kt * 32], a2);
    }
    const float b1 = p.fb_ih[nn], b2 = p.sb_ih[nn];
#pragma unroll
    for (int r = 0; r < 8; ++r) {
      p.giE[(size_t)(mr0 + r) * NG + nn] = a1[r] + b1;
      p.giS[(size_t)(mr0 + r) * NG + nn] = a2[r] + b2;
    }
  }
  gbar(p.bar, ++gen);

  // ---- slow GRU: 64 sequential updates with constant input gates giS
  for (int s = 0; s < NSLOW; ++s) {
    stage_to_lds((s == 0) ? p.hsA0 : p.hsSlots + (size_t)(s - 1) * 65536,
                 ldsA, wave);
    {
      const v16bf* Bm = (const v16bf*)p.SwhhB + (size_t)nt * KT_HH * 32 + lane;
      v8f acc = {};
#pragma unroll 4
      for (int kt = 0; kt < KT_HH; ++kt)
        acc = wmma_bf16(lA[kt * 32], Bm[kt * 32], acc);
      const float bb = p.sb_hh[nn];
#pragma unroll
      for (int r = 0; r < 8; ++r)
        p.gB[(size_t)(mr0 + r) * NG + nn] = acc[r] + bb;
    }
    gbar(p.bar, ++gen);
    __bf16* slot = p.hsSlots + (size_t)s * 65536;
    for (int idx = tid; idx < Bq * HSq; idx += NWG * TPB) {
      const int m = idx >> 10, j = idx & 1023;
      const size_t o = (size_t)m * NG + j;
      const float r   = sigmf(p.giS[o]        + p.gB[o]);
      const float u   = sigmf(p.giS[o + 1024] + p.gB[o + 1024]);
      const float nn2 = tanhf(p.giS[o + 2048] + r * p.gB[o + 2048]);
      const float h2  = (1.0f - u) * nn2 + u * p.hsF[idx];
      p.hsF[idx] = h2;
      const int mt2 = m >> 4, kt2 = j >> 5, kk = j & 31;
      int khalf, jj;
      if (kk < 16) { khalf = kk >> 3; jj = kk & 7; }
      else         { int k2 = kk - 16; khalf = k2 >> 3; jj = 8 + (k2 & 7); }
      const int l2 = khalf * 16 + (m & 15);
      slot[(size_t)(mt2 * KT_HH + kt2) * 512 + (size_t)l2 * 16 + jj] = (__bf16)h2;
    }
    gbar(p.bar, ++gen);
  }

  // ---- fast GRU: 512 sequential steps
  for (int t = 0; t < Tq; ++t) {
    if ((t & 7) == 0) {  // refresh h_slow contribution once per K=8 steps
      stage_to_lds(p.hsSlots + (size_t)(t >> 3) * 65536, ldsA, wave);
      const v16bf* Bm = (const v16bf*)p.WhsB + (size_t)nt * KT_HH * 32 + lane;
      v8f acc = {};
#pragma unroll 4
      for (int kt = 0; kt < KT_HH; ++kt)
        acc = wmma_bf16(lA[kt * 32], Bm[kt * 32], acc);
#pragma unroll
      for (int r = 0; r < 8; ++r) {
        const size_t o = (size_t)(mr0 + r) * NG + nn;
        p.giBase[o] = acc[r] + p.giE[o];
      }
      gbar(p.bar, ++gen);
    }
    // TDM-broadcast current h (A layout) to this WGP's LDS
    stage_to_lds(p.hA, ldsA, wave);
    // hidden-side gates: gB = h @ Whh^T + b_hh    (K=1024, A from LDS)
    {
      const v16bf* Bm = (const v16bf*)p.WhhB + (size_t)nt * KT_HH * 32 + lane;
      v8f acc = {};
#pragma unroll 4
      for (int kt = 0; kt < KT_HH; ++kt)
        acc = wmma_bf16(lA[kt * 32], Bm[kt * 32], acc);
      const float bb = p.fb_hh[nn];
#pragma unroll
      for (int r = 0; r < 8; ++r)
        p.gB[(size_t)(mr0 + r) * NG + nn] = acc[r] + bb;
    }
    // input-side gates: gA = z_prev @ Wz^T + giBase   (K=128 only)
    {
      const v16bf* A  = (const v16bf*)p.ztpA + (size_t)t * 512 +
                        (size_t)mt * KT_Z * 32 + lane;
      const v16bf* Bm = (const v16bf*)p.WzB + (size_t)nt * KT_Z * 32 + lane;
      v8f acc = {};
#pragma unroll
      for (int kt = 0; kt < KT_Z; ++kt)
        acc = wmma_bf16(A[kt * 32], Bm[kt * 32], acc);
#pragma unroll
      for (int r = 0; r < 8; ++r) {
        const size_t o = (size_t)(mr0 + r) * NG + nn;
        p.gA[o] = acc[r] + p.giBase[o];
      }
    }
    gbar(p.bar, ++gen);
    // GRU cell elementwise + repack h into A-layout bf16
    for (int idx = tid; idx < Bq * HFq; idx += NWG * TPB) {
      const int m = idx >> 10, j = idx & 1023;
      const size_t o = (size_t)m * NG + j;
      const float r   = sigmf(p.gA[o]        + p.gB[o]);
      const float u   = sigmf(p.gA[o + 1024] + p.gB[o + 1024]);
      const float nn2 = tanhf(p.gA[o + 2048] + r * p.gB[o + 2048]);
      const float h2  = (1.0f - u) * nn2 + u * p.hF[idx];
      p.hF[idx] = h2;
      const int mt2 = m >> 4, kt2 = j >> 5, kk = j & 31;
      int khalf, jj;
      if (kk < 16) { khalf = kk >> 3; jj = kk & 7; }
      else         { int k2 = kk - 16; khalf = k2 >> 3; jj = 8 + (k2 & 7); }
      const int l2 = khalf * 16 + (m & 15);
      p.hA[(size_t)(mt2 * KT_HH + kt2) * 512 + (size_t)l2 * 16 + jj] = (__bf16)h2;
    }
    gbar(p.bar, ++gen);
    // fused output heads: mu / logvar tiles (64 waves), straight to d_out.
    // (reads the just-updated global hA; LDS copy is one step stale)
    if (gw < 64) {
      const int pmt = gw & 3;
      const int pnt = (gw >> 2) & 7;
      const int sel = gw >> 5;  // 0=mu, 1=logvar
      const v16bf* A  = (const v16bf*)p.hA + (size_t)pmt * KT_HH * 32 + lane;
      const v16bf* Bm = (const v16bf*)(sel ? p.LvB : p.MuB) +
                        (size_t)pnt * KT_HH * 32 + lane;
      v8f acc = {};
#pragma unroll 4
      for (int kt = 0; kt < KT_HH; ++kt)
        acc = wmma_bf16(A[kt * 32], Bm[kt * 32], acc);
      const int zz = pnt * 16 + (lane & 15);
      const float bb = sel ? p.lv_b[zz] : p.mu_b[zz];
      const int pm0 = pmt * 16 + (lane >> 4) * 8;
      float* ob = p.out + (size_t)sel * Bq * ZDq * Tq;
#pragma unroll
      for (int r = 0; r < 8; ++r)
        ob[(size_t)(pm0 + r) * ZDq * Tq + (size_t)zz * Tq + t] = acc[r] + bb;
    }
  }
}

// ---------------- host side ----------------
static inline size_t alup(size_t x) { return (x + 255) & ~(size_t)255; }
static inline int gblocks(size_t total, int cap) {
  size_t g = (total + TPB - 1) / TPB;
  return (int)(g < (size_t)cap ? g : (size_t)cap);
}

extern "C" void kernel_launch(void* const* d_in, const int* in_sizes, int n_in,
                              void* d_out, int out_size, void* d_ws, size_t ws_size,
                              hipStream_t stream) {
  (void)in_sizes; (void)n_in; (void)out_size; (void)ws_size;
  const float* z      = (const float*)d_in[0];
  const float* e      = (const float*)d_in[1];
  const float* s_wih  = (const float*)d_in[2];
  const float* s_whh  = (const float*)d_in[3];
  const float* s_bih  = (const float*)d_in[4];
  const float* s_bhh  = (const float*)d_in[5];
  const float* f_wih  = (const float*)d_in[6];
  const float* f_whh  = (const float*)d_in[7];
  const float* f_bih  = (const float*)d_in[8];
  const float* f_bhh  = (const float*)d_in[9];
  const float* mu_w   = (const float*)d_in[10];
  const float* mu_b   = (const float*)d_in[11];
  const float* lv_w   = (const float*)d_in[12];
  const float* lv_b   = (const float*)d_in[13];
  // d_in[14] = K (==8, compile-time constant here)

  char* w = (char*)d_ws;
  size_t off = 0;
  auto take = [&](size_t bytes) { void* pp = w + off; off = alup(off + bytes); return pp; };

  __bf16* WzB   = (__bf16*)take((size_t)ZDq * NG * 2);
  __bf16* WhsB  = (__bf16*)take((size_t)HSq * NG * 2);
  __bf16* WeB   = (__bf16*)take((size_t)EDq * NG * 2);
  __bf16* WhhB  = (__bf16*)take((size_t)HFq * NG * 2);
  __bf16* SwihB = (__bf16*)take((size_t)EDq * NG * 2);
  __bf16* SwhhB = (__bf16*)take((size_t)HSq * NG * 2);
  __bf16* MuB   = (__bf16*)take((size_t)HFq * ZDq * 2);
  __bf16* LvB   = (__bf16*)take((size_t)HFq * ZDq * 2);
  __bf16* eA    = (__bf16*)take((size_t)Bq * EDq * 2);
  __bf16* ztpA  = (__bf16*)take((size_t)Tq * Bq * ZDq * 2);
  __bf16* hsSl  = (__bf16*)take((size_t)NSLOW * Bq * HSq * 2);
  __bf16* hsA0  = (__bf16*)take((size_t)Bq * HSq * 2);
  __bf16* hA    = (__bf16*)take((size_t)Bq * HFq * 2);
  float*  giE   = (float*) take((size_t)Bq * NG * 4);
  float*  giS   = (float*) take((size_t)Bq * NG * 4);
  float*  giBas = (float*) take((size_t)Bq * NG * 4);
  float*  gA    = (float*) take((size_t)Bq * NG * 4);
  float*  gB    = (float*) take((size_t)Bq * NG * 4);
  float*  hF    = (float*) take((size_t)Bq * HFq * 4);
  float*  hsF   = (float*) take((size_t)Bq * HSq * 4);
  unsigned* bar = (unsigned*)take(256);

  // --- weight / activation packing (bf16, WMMA lane layouts) ---
  msdec_pack_b<<<gblocks(NTG * KT_Z  * 32, 2048), TPB, 0, stream>>>(f_wih, ZDq + HSq + EDq, 0,    ZDq, NG, WzB);
  msdec_pack_b<<<gblocks(NTG * KT_HH * 32, 2048), TPB, 0, stream>>>(f_wih, ZDq + HSq + EDq, ZDq,  HSq, NG, WhsB);
  msdec_pack_b<<<gblocks(NTG * KT_E  * 32, 2048), TPB, 0, stream>>>(f_wih, ZDq + HSq + EDq, ZDq + HSq, EDq, NG, WeB);
  msdec_pack_b<<<gblocks(NTG * KT_HH * 32, 2048), TPB, 0, stream>>>(f_whh, HFq, 0, HFq, NG, WhhB);
  msdec_pack_b<<<gblocks(NTG * KT_E  * 32, 2048), TPB, 0, stream>>>(s_wih, EDq, 0, EDq, NG, SwihB);
  msdec_pack_b<<<gblocks(NTG * KT_HH * 32, 2048), TPB, 0, stream>>>(s_whh, HSq, 0, HSq, NG, SwhhB);
  msdec_pack_b<<<gblocks(8 * KT_HH * 32, 2048), TPB, 0, stream>>>(mu_w, HFq, 0, HFq, ZDq, MuB);
  msdec_pack_b<<<gblocks(8 * KT_HH * 32, 2048), TPB, 0, stream>>>(lv_w, HFq, 0, HFq, ZDq, LvB);
  msdec_pack_a<<<gblocks(MTQ * KT_E * 32, 2048), TPB, 0, stream>>>(e, EDq, EDq, Bq, eA);
  msdec_pack_ztp<<<gblocks((size_t)Tq * 8192, 4096), TPB, 0, stream>>>(z, ztpA);
  msdec_init<<<64, TPB, 0, stream>>>(bar, hF, hsF, (unsigned short*)hA, (unsigned short*)hsA0);

  // --- persistent recurrence (dynamic LDS: one 128KB A blob per WGP) ---
  MSDecParams p;
  p.sb_ih = s_bih; p.sb_hh = s_bhh; p.fb_ih = f_bih; p.fb_hh = f_bhh;
  p.mu_b = mu_b; p.lv_b = lv_b;
  p.WzB = WzB; p.WhsB = WhsB; p.WeB = WeB; p.WhhB = WhhB;
  p.SwihB = SwihB; p.SwhhB = SwhhB; p.MuB = MuB; p.LvB = LvB;
  p.eA = eA; p.ztpA = ztpA;
  p.hA = hA; p.hsA0 = hsA0; p.hsSlots = hsSl;
  p.giE = giE; p.giS = giS; p.giBase = giBas; p.gA = gA; p.gB = gB;
  p.hF = hF; p.hsF = hsF;
  p.out = (float*)d_out; p.bar = bar;

  msdec_main<<<NWG, TPB, ABLOB_BYTES, stream>>>(p);
}
